// MS_MSA_Context_47244640256599
// MI455X (gfx1250) — matrix operand order repaired and checked by reading, not compiled, not run on hardware
//
#include <hip/hip_runtime.h>
#include <hip/hip_bf16.h>
#include <math.h>

// Problem constants (reference: B=2, H=W=256, DIM=256, HEADS=8, DHEAD=32)
#define Bn     2
#define Hh     256
#define Wd     256
#define DIMc   256
#define HEADS  8
#define DHEAD  32
#define Np     65536            // H*W
#define NPIX   (Bn * Np)        // 131072 pixels total

typedef __attribute__((ext_vector_type(16))) _Float16 v16h;
typedef __attribute__((ext_vector_type(8)))  _Float16 v8h;
typedef __attribute__((ext_vector_type(8)))  float    v8f;

// ---------------------------------------------------------------------------
// WMMA helpers (V_WMMA_F32_16X16X32_F16, wave32)
// ---------------------------------------------------------------------------
__device__ __forceinline__ v8f wmma16(v16h a, v16h b, v8f c) {
  return __builtin_amdgcn_wmma_f32_16x16x32_f16(
      /*neg_a=*/false, a, /*neg_b=*/false, b,
      /*c_mod=*/(short)0, c, /*reuse_a=*/false, /*reuse_b=*/false);
}

// A tile (M=16, K=32) from row-major memory: m[row][k], row stride ld.
// lane<16 -> row=lane, K {0-15}; lane>=16 -> row=lane-16, K {16-31}.
__device__ __forceinline__ v16h load_a_h(const _Float16* m, int ld, int lane) {
  const int row = lane & 15;
  const int kh  = (lane < 16) ? 0 : 16;
  const _Float16* p = m + (size_t)row * ld + kh;
  v16h a;
#pragma unroll
  for (int j = 0; j < 16; ++j) a[j] = p[j];
  return a;
}
__device__ __forceinline__ v16h load_a_f32(const float* m, int ld, int lane) {
  const int row = lane & 15;
  const int kh  = (lane < 16) ? 0 : 16;
  const float* p = m + (size_t)row * ld + kh;
  v16h a;
#pragma unroll
  for (int j = 0; j < 16; ++j) a[j] = (_Float16)p[j];
  return a;
}
// B tile (K=32, N=16) where memory holds B^T row-major (rows = output cols,
// contiguous K).  lane%16 = column, K half selected by lane/16.  Two 16B loads.
__device__ __forceinline__ v16h load_bt_h(const _Float16* m, int ld, int lane) {
  const int col = lane & 15;
  const int kb  = (lane < 16) ? 0 : 16;
  const _Float16* p = m + (size_t)col * ld + kb;
  v16h b;
#pragma unroll
  for (int j = 0; j < 16; ++j) b[j] = p[j];
  return b;
}
// B tile (K=32, N=16) from K-major f32 memory m[k][n] (contiguous n), stride ld.
__device__ __forceinline__ v16h load_bk_f32(const float* m, int ld, int lane) {
  const int n  = lane & 15;
  const int kb = (lane < 16) ? 0 : 16;
  v16h b;
#pragma unroll
  for (int j = 0; j < 16; ++j) b[j] = (_Float16)m[(size_t)(kb + j) * ld + n];
  return b;
}

// ---------------------------------------------------------------------------
// Weight pre-conversion kernels (run once per launch; <1.5 MB total)
// ---------------------------------------------------------------------------
// dst[o][c] = (f16) src[c][cols_o + o]   (transpose K-major -> B^T row-major)
__global__ __launch_bounds__(256) void k_cvt_t(const float* __restrict__ src,
                                               _Float16* __restrict__ dst,
                                               int rows_c, int cols_o) {
  const int o = blockIdx.x;        // output row of dst
  const int c = threadIdx.x;       // contiguous in dst
  dst[(size_t)o * rows_c + c] = (_Float16)src[(size_t)c * cols_o + o];
}
// straight f32 -> f16 copy (already [o][c] layout)
__global__ __launch_bounds__(256) void k_cvt(const float* __restrict__ src,
                                             _Float16* __restrict__ dst) {
  const int i = blockIdx.x * 256 + threadIdx.x;
  dst[i] = (_Float16)src[i];
}

// ---------------------------------------------------------------------------
// Kernel 1: mask 3->256 projection + two fused 256x256 conv1x1 GEMMs (WMMA).
// ---------------------------------------------------------------------------
__global__ __launch_bounds__(256) void k_mask_c1c2(
    const float* __restrict__ mask, const float* __restrict__ w_pm,
    const float* __restrict__ b_pm, const _Float16* __restrict__ wc1h,
    const float* __restrict__ b_c1, const _Float16* __restrict__ wc2h,
    const float* __restrict__ b_c2, float* __restrict__ m1,
    float* __restrict__ c2) {
  __shared__ _Float16 ms[16 * 256];
  __shared__ _Float16 c1s[16 * 256];
  const int tid = threadIdx.x, lane = tid & 31, wave = tid >> 5;
  const int b   = blockIdx.x / (Np / 16);
  const int hw0 = (blockIdx.x % (Np / 16)) * 16;
  const size_t p0 = (size_t)b * Np + hw0;
#pragma unroll
  for (int i = 0; i < 16; ++i) {
    const int c = tid;
    const float a0 = mask[(size_t)b * 3 * Np + 0 * Np + hw0 + i];
    const float a1 = mask[(size_t)b * 3 * Np + 1 * Np + hw0 + i];
    const float a2 = mask[(size_t)b * 3 * Np + 2 * Np + hw0 + i];
    const float mv = b_pm[c] + a0 * w_pm[c * 3 + 0] + a1 * w_pm[c * 3 + 1] +
                     a2 * w_pm[c * 3 + 2];
    ms[i * 256 + c] = (_Float16)mv;
  }
  __syncthreads();
  const int prow = ((lane >> 4) << 3);
  const int col  = lane & 15;
  // GEMM1: m @ w_c1^T -> m1 (global f32) + c1s (LDS f16)
  {
    v8f acc[2] = {{}, {}};
    for (int k0 = 0; k0 < 256; k0 += 32) {
      v16h a = load_a_h(ms + k0, 256, lane);
#pragma unroll
      for (int t = 0; t < 2; ++t) {
        v16h bm = load_bt_h(wc1h + (size_t)((wave * 2 + t) * 16) * 256 + k0, 256, lane);
        acc[t] = wmma16(a, bm, acc[t]);
      }
    }
#pragma unroll
    for (int t = 0; t < 2; ++t) {
      const int o = (wave * 2 + t) * 16 + col;
      const float bc = b_c1[o];
#pragma unroll
      for (int i = 0; i < 8; ++i) {
        const int pix = prow + i;
        const float val = acc[t][i] + bc;
        m1[(p0 + pix) * 256 + o] = val;
        c1s[pix * 256 + o] = (_Float16)val;
      }
    }
  }
  __syncthreads();
  // GEMM2: c1 @ w_c2^T -> c2 (global f32)
  {
    v8f acc[2] = {{}, {}};
    for (int k0 = 0; k0 < 256; k0 += 32) {
      v16h a = load_a_h(c1s + k0, 256, lane);
#pragma unroll
      for (int t = 0; t < 2; ++t) {
        v16h bm = load_bt_h(wc2h + (size_t)((wave * 2 + t) * 16) * 256 + k0, 256, lane);
        acc[t] = wmma16(a, bm, acc[t]);
      }
    }
#pragma unroll
    for (int t = 0; t < 2; ++t) {
      const int o = (wave * 2 + t) * 16 + col;
      const float bc = b_c2[o];
#pragma unroll
      for (int i = 0; i < 8; ++i)
        c2[(p0 + prow + i) * 256 + o] = acc[t][i] + bc;
    }
  }
}

// ---------------------------------------------------------------------------
// Kernel 2: 5x5 depthwise conv on c2, sigmoid gate, maskf = m1*a + m1 (in place).
// ---------------------------------------------------------------------------
__global__ __launch_bounds__(256) void k_mask_gate(
    const float* __restrict__ c2, const float* __restrict__ w_dc,
    const float* __restrict__ b_dc, float* __restrict__ m1) {
  const int p  = blockIdx.x;
  const int c  = threadIdx.x;
  const int b  = p >> 16;
  const int hw = p & (Np - 1);
  const int y = hw >> 8, x = hw & 255;
  float acc = b_dc[c];
#pragma unroll
  for (int dy = -2; dy <= 2; ++dy) {
    const int yy = y + dy;
    if (yy < 0 || yy >= Hh) continue;
#pragma unroll
    for (int dx = -2; dx <= 2; ++dx) {
      const int xx = x + dx;
      if (xx < 0 || xx >= Wd) continue;
      acc += c2[((size_t)b * Np + yy * Wd + xx) * 256 + c] *
             w_dc[c * 25 + (dy + 2) * 5 + (dx + 2)];
    }
  }
  const float a = 1.f / (1.f + expf(-acc));
  const float m = m1[(size_t)p * 256 + c];
  m1[(size_t)p * 256 + c] = m * a + m;
}

// ---------------------------------------------------------------------------
// Kernel 3: qkv GEMM (16x768x256, WMMA, f16 pre-transposed weights) + scatter:
//   q,k -> f16 [b][h][d][n]  (packed v8h stores);  v -> f32 [b][hd][n] * maskf
// ---------------------------------------------------------------------------
__global__ __launch_bounds__(256) void k_qkv(
    const float* __restrict__ x, const _Float16* __restrict__ wqkvT,
    const float* __restrict__ maskf, _Float16* __restrict__ q,
    _Float16* __restrict__ k, float* __restrict__ v) {
  __shared__ _Float16 xs[16 * 256];
  const int tid = threadIdx.x, lane = tid & 31, wave = tid >> 5;
  const int b   = blockIdx.x / (Np / 16);
  const int hw0 = (blockIdx.x % (Np / 16)) * 16;
  const size_t p0 = (size_t)b * Np + hw0;
#pragma unroll
  for (int i = 0; i < 16; ++i)
    xs[i * 256 + tid] = (_Float16)x[(p0 + i) * 256 + tid];
  __syncthreads();
  const int prow = ((lane >> 4) << 3);
  const int col  = lane & 15;
  v8f acc[6] = {{}, {}, {}, {}, {}, {}};
  for (int k0 = 0; k0 < 256; k0 += 32) {
    v16h a = load_a_h(xs + k0, 256, lane);
#pragma unroll
    for (int t = 0; t < 6; ++t) {
      const int o0 = (wave * 6 + t) * 16;
      v16h bm = load_bt_h(wqkvT + (size_t)o0 * 256 + k0, 256, lane);
      acc[t] = wmma16(a, bm, acc[t]);
    }
  }
#pragma unroll
  for (int t = 0; t < 6; ++t) {
    const int o0 = (wave * 6 + t) * 16;
    const int o  = o0 + col;
    if (o0 < 256) {            // Q -> [b][h][d][n] f16
      v8h pk;
#pragma unroll
      for (int i = 0; i < 8; ++i) pk[i] = (_Float16)acc[t][i];
      _Float16* dst = q + ((size_t)(b * HEADS + (o >> 5)) * DHEAD + (o & 31)) * Np +
                      hw0 + prow;
      *(v8h*)dst = pk;
    } else if (o0 < 512) {     // K -> [b][h][d][n] f16
      const int ok = o - 256;
      v8h pk;
#pragma unroll
      for (int i = 0; i < 8; ++i) pk[i] = (_Float16)acc[t][i];
      _Float16* dst = k + ((size_t)(b * HEADS + (ok >> 5)) * DHEAD + (ok & 31)) * Np +
                      hw0 + prow;
      *(v8h*)dst = pk;
    } else {                   // V (masked) -> [b][hd][n] f32
      const int ov = o - 512;
      float* dst = v + ((size_t)(b * 256 + ov)) * Np + hw0 + prow;
#pragma unroll
      for (int i = 0; i < 8; ++i)
        dst[i] = acc[t][i] * maskf[(p0 + prow + i) * 256 + ov];
    }
  }
}

// ---------------------------------------------------------------------------
// Kernel 4: per-row (b,h,d) inverse L2 norms over n for q and k.
// ---------------------------------------------------------------------------
__global__ __launch_bounds__(256) void k_rownorm(
    const _Float16* __restrict__ q, const _Float16* __restrict__ k,
    float* __restrict__ qinv, float* __restrict__ kinv) {
  __shared__ float red[256];
  const int r = blockIdx.x;  // 0..511 q rows, 512..1023 k rows
  const bool isq = (r < 512);
  const _Float16* base = (isq ? q : k) + (size_t)(isq ? r : r - 512) * Np;
  float s = 0.f;
  for (int i = threadIdx.x; i < Np; i += 256) {
    const float v = (float)base[i];
    s += v * v;
  }
  red[threadIdx.x] = s;
  __syncthreads();
  for (int off = 128; off > 0; off >>= 1) {
    if (threadIdx.x < off) red[threadIdx.x] += red[threadIdx.x + off];
    __syncthreads();
  }
  if (threadIdx.x == 0) {
    const float inv = 1.f / fmaxf(sqrtf(red[0]), 1e-12f);
    if (isq) qinv[r] = inv; else kinv[r - 512] = inv;
  }
}

// ---------------------------------------------------------------------------
// Kernel 5: Gram matrix G = K·Q^T over n (WMMA) + normalize + softmax.
// One block per (b,h); 8 waves slice n.  Prefetch ~4 iterations ahead.
// ---------------------------------------------------------------------------
__global__ __launch_bounds__(256) void k_attn(
    const _Float16* __restrict__ q, const _Float16* __restrict__ k,
    const float* __restrict__ qinv, const float* __restrict__ kinv,
    const float* __restrict__ rescale, float* __restrict__ attn) {
  __shared__ float Gs[8 * 32 * 32];
  __shared__ float Gf[32 * 32];
  const int tid = threadIdx.x, lane = tid & 31, wave = tid >> 5;
  const int bh = blockIdx.x;          // b*HEADS + h
  const int h  = bh & (HEADS - 1);
  const _Float16* kb = k + (size_t)bh * DHEAD * Np;
  const _Float16* qb = q + (size_t)bh * DHEAD * Np;
  const size_t prowoff = (size_t)(lane & 15) * Np;
  v8f a00 = {}, a01 = {}, a10 = {}, a11 = {};
  for (int n0 = wave * 32; n0 < Np; n0 += 256) {
    // prefetch ~4 iterations ahead (global_prefetch_b8)
    __builtin_prefetch(kb + prowoff + n0 + 1024, 0, 1);
    __builtin_prefetch(kb + prowoff + (size_t)16 * Np + n0 + 1024, 0, 1);
    __builtin_prefetch(qb + prowoff + n0 + 1024, 0, 1);
    __builtin_prefetch(qb + prowoff + (size_t)16 * Np + n0 + 1024, 0, 1);
    v16h A0 = load_a_h(kb + n0, Np, lane);
    v16h A1 = load_a_h(kb + (size_t)16 * Np + n0, Np, lane);
    v16h B0 = load_bt_h(qb + n0, Np, lane);
    v16h B1 = load_bt_h(qb + (size_t)16 * Np + n0, Np, lane);
    a00 = wmma16(A0, B0, a00);
    a01 = wmma16(A0, B1, a01);
    a10 = wmma16(A1, B0, a10);
    a11 = wmma16(A1, B1, a11);
  }
  const int prow = ((lane >> 4) << 3);
  const int col  = lane & 15;
  float* g = Gs + wave * 1024;
#pragma unroll
  for (int i = 0; i < 8; ++i) {
    g[(prow + i) * 32 + col]           = a00[i];
    g[(prow + i) * 32 + 16 + col]      = a01[i];
    g[(16 + prow + i) * 32 + col]      = a10[i];
    g[(16 + prow + i) * 32 + 16 + col] = a11[i];
  }
  __syncthreads();
  const float rs = rescale[h];
#pragma unroll
  for (int j = 0; j < 4; ++j) {
    const int idx = tid + j * 256;
    float s = 0.f;
#pragma unroll
    for (int w = 0; w < 8; ++w) s += Gs[w * 1024 + idx];
    const int d = idx >> 5, e = idx & 31;
    Gf[idx] = s * kinv[bh * 32 + d] * qinv[bh * 32 + e] * rs;
  }
  __syncthreads();
  if (tid < 32) {
    const int d = tid;
    float mx = -1e30f;
    for (int e = 0; e < 32; ++e) mx = fmaxf(mx, Gf[d * 32 + e]);
    float ex[32];
    float sum = 0.f;
    for (int e = 0; e < 32; ++e) { ex[e] = expf(Gf[d * 32 + e] - mx); sum += ex[e]; }
    const float r = 1.f / sum;
    for (int e = 0; e < 32; ++e)
      attn[(size_t)bh * 1024 + d * 32 + e] = ex[e] * r;
  }
}

// ---------------------------------------------------------------------------
// Kernel 6: attn apply (per-head 32x32 @ 32x16, WMMA) fused with output
// projection (16x256 @ 256x256, WMMA, f16 pre-transposed) + bias -> d_out.
// ---------------------------------------------------------------------------
__global__ __launch_bounds__(256) void k_apply_proj(
    const float* __restrict__ attn, const float* __restrict__ v,
    const _Float16* __restrict__ wprojT, const float* __restrict__ b_proj,
    float* __restrict__ out) {
  __shared__ _Float16 ts[16 * 256];   // [pixel][hd]
  const int tid = threadIdx.x, lane = tid & 31, wave = tid >> 5;
  const int b   = blockIdx.x / (Np / 16);
  const int hw0 = (blockIdx.x % (Np / 16)) * 16;
  const int prow = ((lane >> 4) << 3);
  const int col  = lane & 15;
  // stage 1: wave == head; out[d,n] = attn[d,e] @ v[e,n]
  {
    const float* ap = attn + (size_t)(b * HEADS + wave) * 1024;
    const float* vb = v + ((size_t)(b * 256 + wave * 32)) * Np + hw0;
    v16h bm = load_bk_f32(vb, Np, lane);
#pragma unroll
    for (int mt = 0; mt < 2; ++mt) {
      v16h a = load_a_f32(ap + mt * 16 * 32, 32, lane);
      v8f c = {};
      c = wmma16(a, bm, c);
#pragma unroll
      for (int i = 0; i < 8; ++i) {
        const int d = mt * 16 + prow + i;
        ts[col * 256 + wave * 32 + d] = (_Float16)c[i];
      }
    }
  }
  __syncthreads();
  // stage 2: projection with pre-transposed f16 w_proj
  {
    v8f acc[2] = {{}, {}};
    for (int k0 = 0; k0 < 256; k0 += 32) {
      v16h a = load_a_h(ts + k0, 256, lane);
#pragma unroll
      for (int t = 0; t < 2; ++t) {
        const int o0 = (wave * 2 + t) * 16;
        v16h bm = load_bt_h(wprojT + (size_t)o0 * 256 + k0, 256, lane);
        acc[t] = wmma16(a, bm, acc[t]);
      }
    }
#pragma unroll
    for (int t = 0; t < 2; ++t) {
      const int o = (wave * 2 + t) * 16 + col;
      const float bo = b_proj[o];
#pragma unroll
      for (int i = 0; i < 8; ++i)
        out[((size_t)b * Np + hw0 + prow + i) * 256 + o] = acc[t][i] + bo;
    }
  }
}

// ---------------------------------------------------------------------------
// Kernel 7: t = gelu(dwconv3x3(v, pe_w1))  (channel-major, hw-coalesced)
// ---------------------------------------------------------------------------
__global__ __launch_bounds__(256) void k_pos1(
    const float* __restrict__ v, const float* __restrict__ pe_w1,
    float* __restrict__ t) {
  const int cg = blockIdx.x;                 // b*256 + c
  const int c  = cg & 255;
  const int hw = blockIdx.y * 256 + threadIdx.x;
  const int y = hw >> 8, x = hw & 255;
  const float* base = v + (size_t)cg * Np;
  float acc = 0.f;
#pragma unroll
  for (int dy = -1; dy <= 1; ++dy) {
    const int yy = y + dy;
    if (yy < 0 || yy >= Hh) continue;
#pragma unroll
    for (int dx = -1; dx <= 1; ++dx) {
      const int xx = x + dx;
      if (xx < 0 || xx >= Wd) continue;
      acc += base[yy * Wd + xx] * pe_w1[c * 9 + (dy + 1) * 3 + (dx + 1)];
    }
  }
  t[(size_t)cg * Np + hw] =
      0.5f * acc * (1.f + erff(acc * 0.70710678118654752f));
}

// ---------------------------------------------------------------------------
// Kernel 8: d_out += dwconv3x3(t, pe_w2)   (block = pixel, thread = channel)
// ---------------------------------------------------------------------------
__global__ __launch_bounds__(256) void k_pos2(
    const float* __restrict__ t, const float* __restrict__ pe_w2,
    float* __restrict__ out) {
  const int p  = blockIdx.x;
  const int c  = threadIdx.x;
  const int b  = p >> 16;
  const int hw = p & (Np - 1);
  const int y = hw >> 8, x = hw & 255;
  const float* base = t + ((size_t)(b * 256 + c)) * Np;
  float acc = 0.f;
#pragma unroll
  for (int dy = -1; dy <= 1; ++dy) {
    const int yy = y + dy;
    if (yy < 0 || yy >= Hh) continue;
#pragma unroll
    for (int dx = -1; dx <= 1; ++dx) {
      const int xx = x + dx;
      if (xx < 0 || xx >= Wd) continue;
      acc += base[yy * Wd + xx] * pe_w2[c * 9 + (dy + 1) * 3 + (dx + 1)];
    }
  }
  out[(size_t)p * 256 + c] += acc;
}

// ---------------------------------------------------------------------------
extern "C" void kernel_launch(void* const* d_in, const int* in_sizes, int n_in,
                              void* d_out, int out_size, void* d_ws,
                              size_t ws_size, hipStream_t stream) {
  const float* x_in    = (const float*)d_in[0];
  const float* mask    = (const float*)d_in[1];
  const float* w_qkv   = (const float*)d_in[2];
  const float* rescale = (const float*)d_in[3];
  const float* w_proj  = (const float*)d_in[4];
  const float* b_proj  = (const float*)d_in[5];
  const float* pe_w1   = (const float*)d_in[6];
  const float* pe_w2   = (const float*)d_in[7];
  const float* w_pm    = (const float*)d_in[8];
  const float* b_pm    = (const float*)d_in[9];
  const float* w_c1    = (const float*)d_in[10];
  const float* b_c1    = (const float*)d_in[11];
  const float* w_c2    = (const float*)d_in[12];
  const float* b_c2    = (const float*)d_in[13];
  const float* w_dc    = (const float*)d_in[14];
  const float* b_dc    = (const float*)d_in[15];
  float* out = (float*)d_out;

  // workspace carving (all fully written before read each call)
  const size_t FNE = (size_t)NPIX * 256;      // 33,554,432 elements
  float*    m1  = (float*)d_ws;               // m1 / maskf (in place)
  float*    c2  = m1 + FNE;                   // c2, later reused as pos-emb t
  float*    vf  = c2 + FNE;                   // v [b][hd][n]
  _Float16* qh  = (_Float16*)(vf + FNE);      // q f16 [b][h][d][n]
  _Float16* kh  = qh + FNE;                   // k f16
  float*    qinv = (float*)(kh + FNE);        // 512
  float*    kinv = qinv + 512;                // 512
  float*    attn = kinv + 512;                // 16*32*32
  _Float16* wqkvT = (_Float16*)(attn + 16 * 1024);  // [768][256] f16
  _Float16* wc1h  = wqkvT + (size_t)768 * 256;      // [256][256] f16
  _Float16* wc2h  = wc1h + (size_t)256 * 256;
  _Float16* wprojT = wc2h + (size_t)256 * 256;      // [256][256] f16 (transposed)

  const int PB = NPIX / 16;                   // 8192 pixel-tile blocks

  // one-time weight conversion (f32 -> f16, B^T row-major layouts)
  k_cvt_t<<<768, 256, 0, stream>>>(w_qkv, wqkvT, 256, 768);
  k_cvt_t<<<256, 256, 0, stream>>>(w_proj, wprojT, 256, 256);
  k_cvt<<<256, 256, 0, stream>>>(w_c1, wc1h);
  k_cvt<<<256, 256, 0, stream>>>(w_c2, wc2h);

  k_mask_c1c2<<<PB, 256, 0, stream>>>(mask, w_pm, b_pm, wc1h, b_c1, wc2h, b_c2,
                                      m1, c2);
  k_mask_gate<<<NPIX, 256, 0, stream>>>(c2, w_dc, b_dc, m1);
  k_qkv<<<PB, 256, 0, stream>>>(x_in, wqkvT, m1, qh, kh, vf);
  k_rownorm<<<1024, 256, 0, stream>>>(qh, kh, qinv, kinv);
  k_attn<<<Bn * HEADS, 256, 0, stream>>>(qh, kh, qinv, kinv, rescale, attn);
  k_apply_proj<<<PB, 256, 0, stream>>>(attn, vf, wprojT, b_proj, out);
  k_pos1<<<dim3(Bn * 256, Np / 256), 256, 0, stream>>>(vf, pe_w1, c2);
  k_pos2<<<NPIX, 256, 0, stream>>>(c2, pe_w2, out);
}
